// SCOPENet_58119497450384
// MI455X (gfx1250) — compile-verified, matrix operand order checked by reading, not run on hardware
//
#include <hip/hip_runtime.h>
#include <hip/hip_bf16.h>
#include <math.h>

// ---------------------------------------------------------------------------
// SCOPENet deformable-conv block on gfx1250 (CDNA5, wave32, WMMA).
//   B=4, C=64, H=W=128, N=9 (3x3), K = C*9 = 576.
// Implicit-GEMM formulation, v_wmma_f32_16x16x32_f16 fed by ds_load_b128:
//   A fragment = 2 contiguous 16B runs in K  (ISA 7.12.2 16-bit A layout)
//   B fragment = pre-swizzled per-lane 32B records (prep kernel)
// Weight staging uses gfx1250 async global->LDS (ASYNCcnt) when available.
// Everything fits in L2 (x 16MB, om 8MB, out 16MB) -> HBM floor ~2us @23.3TB/s.
// ---------------------------------------------------------------------------

#define BB 4
#define CC 64
#define HH 128
#define WW 128
#define NN 9
#define KK 576            // C * 9
#define KSTEPS (KK / 32)  // 18
#define NB1 2             // 32 out channels (offset+mask, padded)
#define NB2 4             // 64 out channels (final conv)

typedef _Float16 v16h __attribute__((ext_vector_type(16)));
typedef float    v8f  __attribute__((ext_vector_type(8)));
typedef int      v4i  __attribute__((ext_vector_type(4)));

union Frag16 { uint4 q[2]; v16h h; };

// 16-bit A-matrix 16x32 per-lane K index (ISA 7.12.2):
//   lane = 16*half + m ; element e -> K = 16*(e>=8) + 8*half + (e&7)
__device__ __forceinline__ int kmap(int e, int half) {
    return ((e & 8) << 1) + (half << 3) + (e & 7);
}

__device__ __forceinline__ float sigmoidf_(float t) {
    return 1.0f / (1.0f + __expf(-t));
}

// --- async global->LDS staging with graceful fallback ----------------------
#if defined(__has_builtin)
#if __has_builtin(__builtin_amdgcn_global_load_async_to_lds_b128)
#define CDNA5_ASYNC_LDS 1
#endif
#endif

typedef __attribute__((address_space(1))) v4i gv4i_t;   // global v4i
typedef __attribute__((address_space(3))) v4i lv4i_t;   // LDS v4i

__device__ __forceinline__ void stage_to_lds(const void* gsrc, void* ldst,
                                             int bytes, int tid, int nthr) {
#if defined(CDNA5_ASYNC_LDS)
    gv4i_t* g = (gv4i_t*)gsrc;
    lv4i_t* l = (lv4i_t*)ldst;
    for (int i = tid; i < bytes / 16; i += nthr)
        __builtin_amdgcn_global_load_async_to_lds_b128(g + i, l + i, 0, 0);
#if __has_builtin(__builtin_amdgcn_s_wait_asynccnt)
    __builtin_amdgcn_s_wait_asynccnt(0);
#else
    asm volatile("s_wait_asynccnt 0x0" ::: "memory");
#endif
#else
    const uint4* g = (const uint4*)gsrc;
    uint4* l = (uint4*)ldst;
    for (int i = tid; i < bytes / 16; i += nthr) l[i] = g[i];
#endif
}

// ---------------------------------------------------------------------------
// Kernel P: repack weights into WMMA-fragment-ordered f16 B matrices.
//   wpmF  [kb][nb<2][lane<32][e<16]  ch 0..17 = p_w, 18..26 = m_w, rest 0
//   wcnvF [kb][nb<4][lane<32][e<16]  from conv_w[o][c][kh][kw], k = c*9+n
// ---------------------------------------------------------------------------
__global__ void prep_weights(const float* __restrict__ p_w,
                             const float* __restrict__ m_w,
                             const float* __restrict__ conv_w,
                             _Float16* __restrict__ wpmF,
                             _Float16* __restrict__ wcnvF) {
    const int i = blockIdx.x * 256 + threadIdx.x;
    if (i < KSTEPS * NB1 * 32 * 16) {
        const int e = i & 15, lane = (i >> 4) & 31;
        const int nb = (i >> 9) & (NB1 - 1), kb = i >> 10;
        const int k  = kb * 32 + kmap(e, lane >> 4);
        const int ch = nb * 16 + (lane & 15);
        const int c = k / NN, n = k % NN;
        float v = 0.0f;
        if (ch < 18)      v = p_w[(ch * CC + c) * NN + n];
        else if (ch < 27) v = m_w[((ch - 18) * CC + c) * NN + n];
        wpmF[i] = (_Float16)v;
    }
    if (i < KSTEPS * NB2 * 32 * 16) {
        const int e = i & 15, lane = (i >> 4) & 31;
        const int nb = (i >> 9) & (NB2 - 1), kb = i >> 11;
        const int k = kb * 32 + kmap(e, lane >> 4);
        const int o = nb * 16 + (lane & 15);
        wcnvF[i] = (_Float16)conv_w[(o * CC + (k / NN)) * NN + (k % NN)];
    }
}

// ---------------------------------------------------------------------------
// Kernel 1: offset/mask conv as implicit GEMM (M=65536, K=576, N=32).
// Tile: 32 pixels x 32 channels; 4 waves x one 16x16 WMMA tile; 18 K-steps.
// A tile (32x576 f16) built in LDS with b128 stores; frags = 2x ds_load_b128.
// ---------------------------------------------------------------------------
__global__ __launch_bounds__(128) void offmask_gemm(
        const float* __restrict__ x,
        const _Float16* __restrict__ wpmF,
        const float* __restrict__ p_b,
        const float* __restrict__ m_b,
        float* __restrict__ om) {
    __shared__ __align__(16) _Float16 sW[KSTEPS * NB1 * 32 * 16];  // 36 KB
    __shared__ __align__(16) _Float16 sA[32 * KK];                 // 36 KB

    const int tid = threadIdx.x;
    stage_to_lds(wpmF, sW, KSTEPS * NB1 * 32 * 16 * 2, tid, 128);

    const int tilesPerRow = WW / 32;                 // 4
    const int wg = blockIdx.x;                       // 0..2047
    const int w0 = (wg % tilesPerRow) * 32;
    const int h  = (wg / tilesPerRow) % HH;
    const int b  = wg / (tilesPerRow * HH);

    // ---- build A tile: 32 pixels x 36 chunks of 16 consecutive k ----------
    for (int j = tid; j < 32 * (KK / 16); j += 128) {
        const int p = j / (KK / 16), kc = j % (KK / 16);
        const int k0 = kc * 16;
        int c = k0 / NN, n = k0 % NN;
        const int wcol = w0 + p;
        __align__(16) _Float16 tmp[16];
#pragma unroll
        for (int e = 0; e < 16; ++e) {
            const int kh = n / 3, kw = n % 3;
            const int y  = h + kh - 1;
            const int xc = wcol + kw - 1;
            float v = 0.0f;
            if ((unsigned)y < (unsigned)HH && (unsigned)xc < (unsigned)WW)
                v = x[((b * CC + c) * HH + y) * WW + xc];
            tmp[e] = (_Float16)v;
            if (++n == NN) { n = 0; ++c; }
        }
        *(uint4*)&sA[p * KK + k0]     = *(const uint4*)&tmp[0];
        *(uint4*)&sA[p * KK + k0 + 8] = *(const uint4*)&tmp[8];
    }
    __syncthreads();

    const int wave = tid >> 5, lane = tid & 31;
    const int mi = wave >> 1, ni = wave & 1;
    const int half = lane >> 4, lm = lane & 15;

    const _Float16* aRow = &sA[(mi * 16 + lm) * KK + 8 * half];
    const _Float16* bPtr = &sW[(ni * 32 + lane) * 16];

    v8f acc = {};
    for (int kb = 0; kb < KSTEPS; ++kb) {
        Frag16 fa, fb;
        fa.q[0] = *(const uint4*)(aRow + kb * 32);
        fa.q[1] = *(const uint4*)(aRow + kb * 32 + 16);
        fb.q[0] = *(const uint4*)(bPtr + kb * (NB1 * 32 * 16));
        fb.q[1] = *(const uint4*)(bPtr + kb * (NB1 * 32 * 16) + 8);
        acc = __builtin_amdgcn_wmma_f32_16x16x32_f16(
                false, fa.h, false, fb.h, (short)0, acc, false, false);
    }

    // D layout: lane -> (n = lm, m = r + 8*half); fused bias / sigmoid
    const int ch = ni * 16 + lm;
#pragma unroll
    for (int r = 0; r < 8; ++r) {
        const int pw = w0 + mi * 16 + 8 * half + r;
        float v = acc[r];
        if (ch < 18)      v += p_b[ch];
        else if (ch < 27) v = sigmoidf_(v + m_b[ch - 18]);
        else              v = 0.0f;
        om[(((size_t)(b * HH + h) * WW + pw) << 5) + ch] = v;
    }
}

// ---------------------------------------------------------------------------
// Kernel 2: fused deformable sampling + final conv GEMM (M=65536,K=576,N=64).
// Tile: 32 pixels x 64 out channels, 8 waves (2x4 grid of 16x16 WMMA tiles).
// Tap validity folded into bilinear weights -> unconditional gathers.
// ---------------------------------------------------------------------------
__global__ __launch_bounds__(256) void deform_gemm(
        const float* __restrict__ x,
        const _Float16* __restrict__ wcnvF,
        const float* __restrict__ om,
        float* __restrict__ out) {
    __shared__ __align__(16) _Float16 sW[KSTEPS * NB2 * 32 * 16];  // 72 KB
    __shared__ __align__(16) _Float16 sA[32 * KK];                 // 36 KB

    const int tid = threadIdx.x;
    stage_to_lds(wcnvF, sW, KSTEPS * NB2 * 32 * 16 * 2, tid, 256);

    const int tilesPerRow = WW / 32;
    const int wg = blockIdx.x;
    const int w0 = (wg % tilesPerRow) * 32;
    const int h  = (wg / tilesPerRow) % HH;
    const int b  = wg / (tilesPerRow * HH);

    // ---- Phase A: 32x576 x_off tile via deformable bilinear sampling ------
    const float HpM = (float)(HH + 1);               // 129 = Hp-1
    const float WpM = (float)(WW + 1);
    for (int job = tid; job < 32 * NN; job += 256) {
        const int p = job / NN, n = job % NN;
        const int wcol = w0 + p;
        const float* omp = &om[((size_t)(b * HH + h) * WW + wcol) << 5];
        const float offx = omp[n];
        const float offy = omp[NN + n];
        const float mk   = omp[18 + n];
        const int kh = n / 3, kw = n % 3;
        const float px = offx + (float)(kh - 1) + (float)(h + 1);
        const float py = offy + (float)(kw - 1) + (float)(wcol + 1);
        const float qfx = floorf(px), qfy = floorf(py);
        const int qx0 = (int)fminf(fmaxf(qfx,       0.f), HpM);
        const int qy0 = (int)fminf(fmaxf(qfy,       0.f), WpM);
        const int qx1 = (int)fminf(fmaxf(qfx + 1.f, 0.f), HpM);
        const int qy1 = (int)fminf(fmaxf(qfy + 1.f, 0.f), WpM);
        const float pxc = fminf(fmaxf(px, 0.f), HpM);
        const float pyc = fminf(fmaxf(py, 0.f), WpM);
        float glt = (1.f + ((float)qx0 - pxc)) * (1.f + ((float)qy0 - pyc));
        float grb = (1.f - ((float)qx1 - pxc)) * (1.f - ((float)qy1 - pyc));
        float glb = (1.f + ((float)qx0 - pxc)) * (1.f - ((float)qy1 - pyc));
        float grt = (1.f - ((float)qx1 - pxc)) * (1.f + ((float)qy0 - pyc));
        // fold pad-zero taps into the weights; clamp indices in-bounds
        const bool vx0 = (qx0 >= 1) & (qx0 <= HH);
        const bool vx1 = (qx1 >= 1) & (qx1 <= HH);
        const bool vy0 = (qy0 >= 1) & (qy0 <= WW);
        const bool vy1 = (qy1 >= 1) & (qy1 <= WW);
        glt = (vx0 & vy0) ? glt * mk : 0.f;
        grb = (vx1 & vy1) ? grb * mk : 0.f;
        glb = (vx0 & vy1) ? glb * mk : 0.f;
        grt = (vx1 & vy0) ? grt * mk : 0.f;
        const int ix0 = min(max(qx0 - 1, 0), HH - 1);
        const int ix1 = min(max(qx1 - 1, 0), HH - 1);
        const int iy0 = min(max(qy0 - 1, 0), WW - 1);
        const int iy1 = min(max(qy1 - 1, 0), WW - 1);
        const int r00 = ix0 * WW + iy0;
        const int r11 = ix1 * WW + iy1;
        const int r01 = ix0 * WW + iy1;
        const int r10 = ix1 * WW + iy0;
        const float* xb = x + (size_t)b * CC * HH * WW;
        __builtin_prefetch(xb + r00, 0, 0);          // global_prefetch_b8
        _Float16* sdst = &sA[p * KK + n];
        for (int c = 0; c < CC; ++c) {
            const float* xc = xb + (size_t)c * HH * WW;
            const float val = glt * xc[r00] + grb * xc[r11] +
                              glb * xc[r01] + grt * xc[r10];
            sdst[c * NN] = (_Float16)val;
        }
    }
    __syncthreads();

    // ---- Phase B: 32x64 output tile via WMMA -------------------------------
    const int wave = tid >> 5, lane = tid & 31;
    const int mi = wave >> 2, ni = wave & 3;
    const int half = lane >> 4, lm = lane & 15;

    const _Float16* aRow = &sA[(mi * 16 + lm) * KK + 8 * half];
    const _Float16* bPtr = &sW[(ni * 32 + lane) * 16];

    v8f acc = {};
    for (int kb = 0; kb < KSTEPS; ++kb) {
        Frag16 fa, fb;
        fa.q[0] = *(const uint4*)(aRow + kb * 32);
        fa.q[1] = *(const uint4*)(aRow + kb * 32 + 16);
        fb.q[0] = *(const uint4*)(bPtr + kb * (NB2 * 32 * 16));
        fb.q[1] = *(const uint4*)(bPtr + kb * (NB2 * 32 * 16) + 8);
        acc = __builtin_amdgcn_wmma_f32_16x16x32_f16(
                false, fa.h, false, fb.h, (short)0, acc, false, false);
    }

    // D: lane -> (o = ni*16+lm, pixels pwBase..pwBase+7) -> two b128 stores
    const int o = ni * 16 + lm;
    const int pwBase = w0 + mi * 16 + 8 * half;
    float* op = &out[(((size_t)b * CC + o) * HH + h) * WW + pwBase];
    float4 lo4 = make_float4(acc[0], acc[1], acc[2], acc[3]);
    float4 hi4 = make_float4(acc[4], acc[5], acc[6], acc[7]);
    *(float4*)op       = lo4;
    *(float4*)(op + 4) = hi4;
}

// ---------------------------------------------------------------------------
// Launch: inputs in setup_inputs() order: x, p_w, p_b, m_w, m_b, conv_w.
// Workspace: [0, 8MB) om (B*H*W x 32 f32), then fragment-ordered f16 weights.
// ---------------------------------------------------------------------------
extern "C" void kernel_launch(void* const* d_in, const int* in_sizes, int n_in,
                              void* d_out, int out_size, void* d_ws, size_t ws_size,
                              hipStream_t stream) {
    const float* x      = (const float*)d_in[0];
    const float* p_w    = (const float*)d_in[1];
    const float* p_b    = (const float*)d_in[2];
    const float* m_w    = (const float*)d_in[3];
    const float* m_b    = (const float*)d_in[4];
    const float* conv_w = (const float*)d_in[5];
    float* out = (float*)d_out;

    char* ws = (char*)d_ws;
    float*    om    = (float*)ws;                                   // 8 MB
    _Float16* wpmF  = (_Float16*)(ws + (size_t)8 * 1024 * 1024);    // 36 KB
    _Float16* wcnvF = (_Float16*)(ws + (size_t)8 * 1024 * 1024 + 64 * 1024); // 72 KB

    prep_weights<<<(KSTEPS * NB2 * 32 * 16 + 255) / 256, 256, 0, stream>>>(
        p_w, m_w, conv_w, wpmF, wcnvF);

    const int tiles = BB * HH * (WW / 32);           // 2048
    offmask_gemm<<<tiles, 128, 0, stream>>>(x, wpmF, p_b, m_b, om);
    deform_gemm<<<tiles, 256, 0, stream>>>(x, wcnvF, om, out);
}
